// MLA_49615462204150
// MI455X (gfx1250) — matrix operand order repaired and checked by reading, not compiled, not run on hardware
//
#include <hip/hip_runtime.h>

// ---------------------------------------------------------------------------
// MLA forward for MI455X (gfx1250, wave32, WMMA).
// Pipeline: fp32->bf16 convert/transpose -> WMMA GEMMs -> register-resident
// flash attention (transposed-score trick, no LDS) -> WMMA output GEMM (fp32).
// ---------------------------------------------------------------------------

typedef unsigned short u16;
typedef __bf16        v16bf __attribute__((ext_vector_type(16)));
typedef float         v8f   __attribute__((ext_vector_type(8)));
typedef unsigned int  v4u   __attribute__((ext_vector_type(4)));

#define B_  2
#define L_  2048
#define D_  1024
#define H_  16
#define HD_ 64
#define LD_ 256
#define NQ_ (H_ * HD_)   // 1024

union FragBF {
  v16bf v;
  v4u   u4[2];
  unsigned int   u[8];
  unsigned short s[16];
};

__device__ __forceinline__ unsigned short f2bf(float f) {
  unsigned int u = __float_as_uint(f);
  return (unsigned short)((u + 0x7FFFu + ((u >> 16) & 1u)) >> 16);  // RNE
}

__device__ __forceinline__ v8f wmma_bf16(const FragBF& a, const FragBF& b, v8f c) {
  // D = A(16x32) * B(32x16) + C, fp32 accumulate
  return __builtin_amdgcn_wmma_f32_16x16x32_bf16(
      false, a.v, false, b.v, (short)0, c, false, false);
}

// A-operand fragment: 16x32 bf16. Row m = lane&15. K order:
// element e <-> K = e + 16*half (matches load_fragB).
// Loads two b128s (K-pairs are contiguous in row-major [M][K] memory).
__device__ __forceinline__ void load_fragA(FragBF& f, const u16* base, int ld, int lane) {
  int m = lane & 15, hh = lane >> 4;
  const u16* p = base + (size_t)m * ld + hh * 16;
  f.u4[0] = *(const v4u*)(p);
  f.u4[1] = *(const v4u*)(p + 8);
}

// B-operand fragment: 32x16 bf16, loaded from row-major B^T [n][k].
// Column n = lane&15; lanes 0-15 hold K=0..15, lanes 16-31 hold K=16..31
// (element e <-> k = e + 16*half). Two b128 loads.
__device__ __forceinline__ void load_fragB(FragBF& f, const u16* base, int ld, int lane) {
  int n = lane & 15, hh = lane >> 4;
  const u16* p = base + (size_t)n * ld + hh * 16;
  f.u4[0] = *(const v4u*)(p);
  f.u4[1] = *(const v4u*)(p + 8);
}

__device__ __forceinline__ v4u pack8(v8f a, float scale) {
  v4u w;
#pragma unroll
  for (int d = 0; d < 4; ++d) {
    unsigned int lo = f2bf(a[2 * d] * scale);
    unsigned int hi = f2bf(a[2 * d + 1] * scale);
    w[d] = lo | (hi << 16);
  }
  return w;
}

// ---------------------------------------------------------------------------
// Elementwise fp32 -> bf16
__global__ void k_cvt(const float* __restrict__ src, u16* __restrict__ dst, int n) {
  int idx = blockIdx.x * blockDim.x + threadIdx.x;
  if (idx < n) dst[idx] = f2bf(src[idx]);
}

// Transpose + convert: src fp32 [K][N] row-major -> dst bf16 [N][K]
__global__ void k_tcvt(const float* __restrict__ src, u16* __restrict__ dst, int K, int N) {
  int idx = blockIdx.x * blockDim.x + threadIdx.x;
  if (idx < K * N) {
    int k = idx / N, n = idx % N;
    dst[(size_t)n * K + k] = f2bf(src[idx]);
  }
}

// ---------------------------------------------------------------------------
// GEMM: C[M][N] = A[M][K] * Bt[N][K]^T, bf16 inputs, fp32 accumulate.
// One wave computes a 16x64 tile (A fragment reused across 4 B tiles).
// mode 0: bf16 row-major out (with scale)   mode 1: bf16 v^T out [B][H][HD][L]
// mode 2: fp32 row-major out
__global__ __launch_bounds__(32) void k_gemm_nt(const u16* __restrict__ A,
                                                const u16* __restrict__ Bt,
                                                void* __restrict__ out,
                                                int M, int N, int K,
                                                float scale, int mode) {
  int lane = threadIdx.x;
  int col0 = blockIdx.x * 64;
  int row0 = blockIdx.y * 16;
  int nl = lane & 15, hh = lane >> 4;

  v8f acc0 = {}, acc1 = {}, acc2 = {}, acc3 = {};
  const u16* Arow = A + (size_t)row0 * K;
  const u16* Bp0 = Bt + (size_t)(col0 + 0) * K;
  const u16* Bp1 = Bt + (size_t)(col0 + 16) * K;
  const u16* Bp2 = Bt + (size_t)(col0 + 32) * K;
  const u16* Bp3 = Bt + (size_t)(col0 + 48) * K;

  for (int k0 = 0; k0 < K; k0 += 32) {
    FragBF a, b;
    load_fragA(a, Arow + k0, K, lane);
    __builtin_prefetch(Arow + k0 + 32, 0, 3);   // global_prefetch_b8
    load_fragB(b, Bp0 + k0, K, lane); acc0 = wmma_bf16(a, b, acc0);
    load_fragB(b, Bp1 + k0, K, lane); acc1 = wmma_bf16(a, b, acc1);
    load_fragB(b, Bp2 + k0, K, lane); acc2 = wmma_bf16(a, b, acc2);
    load_fragB(b, Bp3 + k0, K, lane); acc3 = wmma_bf16(a, b, acc3);
  }
  v8f acc[4] = {acc0, acc1, acc2, acc3};

  if (mode == 0) {
    u16* o = (u16*)out;
#pragma unroll
    for (int t = 0; t < 4; ++t) {
      int n = col0 + t * 16 + nl;
#pragma unroll
      for (int r = 0; r < 8; ++r) {
        int m = row0 + r + 8 * hh;   // C rows: m = vgpr + 8*half
        o[(size_t)m * N + n] = f2bf(acc[t][r] * scale);
      }
    }
  } else if (mode == 1) {  // v^T: [B][H][HD][L]; requires N==1024, M==B*L
    u16* o = (u16*)out;
    int bb = row0 >> 11;                 // row / L
    int lb = (row0 & (L_ - 1)) + 8 * hh; // sequence position base (contiguous over r)
#pragma unroll
    for (int t = 0; t < 4; ++t) {
      int n = col0 + t * 16 + nl;
      int h = n >> 6, hd = n & 63;
      u16* dst = o + (((size_t)(bb * H_ + h) * HD_ + hd) * L_ + lb);
      *(v4u*)dst = pack8(acc[t], scale);  // 8 consecutive l positions, b128 store
    }
  } else {  // fp32 row-major
    float* o = (float*)out;
#pragma unroll
    for (int t = 0; t < 4; ++t) {
      int n = col0 + t * 16 + nl;
#pragma unroll
      for (int r = 0; r < 8; ++r) {
        int m = row0 + r + 8 * hh;
        o[(size_t)m * N + n] = acc[t][r] * scale;
      }
    }
  }
}

// ---------------------------------------------------------------------------
// Flash attention, one wave per (b, h, 16-query tile), all in registers.
// Scores computed transposed: S'[j][i] = K * Q^T  so queries live in lanes;
// output accumulated transposed: O'[hd][i] = V^T * P'. Online softmax is
// per-lane + one xor-16 shuffle; no LDS, no block barriers.
__global__ __launch_bounds__(32) void k_flash(const u16* __restrict__ qb,
                                              const u16* __restrict__ kb,
                                              const u16* __restrict__ vT,
                                              u16* __restrict__ yb) {
  int b = blockIdx.z, h = blockIdx.y;
  int qi0 = blockIdx.x * 16;
  int lane = threadIdx.x;
  int i = lane & 15, hh = lane >> 4;
  int iq = qi0 + i;  // this lane's query index

  FragBF fq0, fq1;  // Q as B-operand, hd K-steps 0..31 and 32..63 (scale folded in qb)
  const u16* qbase = qb + ((size_t)(b * L_ + qi0)) * NQ_ + h * HD_;
  load_fragB(fq0, qbase, NQ_, lane);
  load_fragB(fq1, qbase + 32, NQ_, lane);

  const u16* kbase = kb + (size_t)b * L_ * NQ_ + h * HD_;
  const u16* vbase = vT + ((size_t)(b * H_ + h)) * HD_ * L_;

  float m_i = -1e30f, l_i = 0.0f;
  v8f o0 = {}, o1 = {}, o2 = {}, o3 = {};  // O'[hd][i], 4 hd row-tiles

  for (int j0 = 0; j0 <= qi0; j0 += 32) {  // causal: last tile partially masked
    v8f c0 = {}, c1 = {};
    FragBF ak;
    load_fragA(ak, kbase + (size_t)j0 * NQ_, NQ_, lane);            c0 = wmma_bf16(ak, fq0, c0);
    load_fragA(ak, kbase + (size_t)j0 * NQ_ + 32, NQ_, lane);       c0 = wmma_bf16(ak, fq1, c0);
    load_fragA(ak, kbase + (size_t)(j0 + 16) * NQ_, NQ_, lane);     c1 = wmma_bf16(ak, fq0, c1);
    load_fragA(ak, kbase + (size_t)(j0 + 16) * NQ_ + 32, NQ_, lane);c1 = wmma_bf16(ak, fq1, c1);

    // causal mask + running max (C rows: key j = j0 + r + 8*half [+16 for c1])
    float s0[8], s1[8];
    float mloc = -1e30f;
#pragma unroll
    for (int r = 0; r < 8; ++r) {
      int j = j0 + r + 8 * hh;
      s0[r] = (j <= iq) ? c0[r] : -1e30f;
      s1[r] = (j + 16 <= iq) ? c1[r] : -1e30f;
      mloc = fmaxf(mloc, fmaxf(s0[r], s1[r]));
    }
    mloc = fmaxf(mloc, __shfl_xor(mloc, 16, 32));  // combine key-halves
    float m_new = fmaxf(m_i, mloc);
    float corr = __expf(m_i - m_new);

    float p0[8], p1[8], psum = 0.f;
#pragma unroll
    for (int r = 0; r < 8; ++r) {
      p0[r] = __expf(s0[r] - m_new);
      p1[r] = __expf(s1[r] - m_new);
      psum += p0[r] + p1[r];
    }
    psum += __shfl_xor(psum, 16, 32);
    l_i = l_i * corr + psum;
    m_i = m_new;
#pragma unroll
    for (int r = 0; r < 8; ++r) { o0[r] *= corr; o1[r] *= corr; o2[r] *= corr; o3[r] *= corr; }

    // Build P' B-operand: element e of lane holds k = j0 + e + 16*half.
    // C-fragment rows interleave by 8, so swap 8-row halves via xor-16 shuffle.
    float q0[8], q1[8];
#pragma unroll
    for (int r = 0; r < 8; ++r) {
      q0[r] = __shfl_xor(p0[r], 16, 32);
      q1[r] = __shfl_xor(p1[r], 16, 32);
    }
    FragBF pf;
#pragma unroll
    for (int e = 0; e < 8; ++e) {
      float lo = hh ? q1[e] : p0[e];  // k = e + 16*half
      float hi = hh ? p1[e] : q0[e];  // k = 8 + e + 16*half
      pf.s[e] = f2bf(lo);
      pf.s[8 + e] = f2bf(hi);
    }

    // O' += V^T * P' : A = vT rows hd (contiguous along key j), B = P'
    FragBF av;
    load_fragA(av, vbase + (size_t)0 * 16 * L_ + j0, L_, lane); o0 = wmma_bf16(av, pf, o0);
    load_fragA(av, vbase + (size_t)1 * 16 * L_ + j0, L_, lane); o1 = wmma_bf16(av, pf, o1);
    load_fragA(av, vbase + (size_t)2 * 16 * L_ + j0, L_, lane); o2 = wmma_bf16(av, pf, o2);
    load_fragA(av, vbase + (size_t)3 * 16 * L_ + j0, L_, lane); o3 = wmma_bf16(av, pf, o3);
  }

  // Normalize (per-lane, since lanes own queries) and store y[b,l,h*64+hd] bf16.
  float inv = 1.0f / l_i;
  u16* yp = yb + ((size_t)(b * L_ + qi0 + i)) * NQ_ + h * HD_ + 8 * hh;
  *(v4u*)(yp + 0)  = pack8(o0, inv);   // hd = 0..7   (+8*half)
  *(v4u*)(yp + 16) = pack8(o1, inv);   // hd = 16..23 (+8*half)
  *(v4u*)(yp + 32) = pack8(o2, inv);
  *(v4u*)(yp + 48) = pack8(o3, inv);
}

// ---------------------------------------------------------------------------
extern "C" void kernel_launch(void* const* d_in, const int* in_sizes, int n_in,
                              void* d_out, int out_size, void* d_ws, size_t ws_size,
                              hipStream_t stream) {
  (void)in_sizes; (void)n_in; (void)out_size; (void)ws_size;
  const float* x    = (const float*)d_in[0];  // [B,L,D]
  const float* Wq   = (const float*)d_in[1];  // [D,H,HD]
  const float* Wkvd = (const float*)d_in[2];  // [D,LD]
  const float* Wkup = (const float*)d_in[3];  // [LD,H,HD]
  const float* Wvup = (const float*)d_in[4];  // [LD,H,HD]
  const float* Wout = (const float*)d_in[5];  // [H,HD,D]
  float* out = (float*)d_out;                 // [B,L,D] fp32

  const int M = B_ * L_;  // 4096
  char* ws = (char*)d_ws;
  size_t off = 0;
  auto alloc = [&](size_t elems) { u16* p = (u16*)(ws + off); off += elems * sizeof(u16); return p; };

  u16* xb    = alloc((size_t)M * D_);      // x bf16 [M][D]
  u16* WqT   = alloc((size_t)NQ_ * D_);    // [N=1024][K=1024]
  u16* WkvdT = alloc((size_t)LD_ * D_);    // [256][1024]
  u16* WkupT = alloc((size_t)NQ_ * LD_);   // [1024][256]
  u16* WvupT = alloc((size_t)NQ_ * LD_);   // [1024][256]
  u16* WoutT = alloc((size_t)D_ * NQ_);    // [1024][1024]
  u16* qb    = alloc((size_t)M * NQ_);     // q bf16 (pre-scaled by 1/8)
  u16* lat   = alloc((size_t)M * LD_);     // latent bf16
  u16* kb    = alloc((size_t)M * NQ_);     // k bf16 [M][H*HD]
  u16* vT    = alloc((size_t)B_ * H_ * HD_ * L_);  // v transposed [B][H][HD][L]
  u16* yb    = alloc((size_t)M * NQ_);     // attention output bf16
  // total ~47.5 MB of workspace

  int n;
  n = M * D_;
  k_cvt<<<(n + 255) / 256, 256, 0, stream>>>(x, xb, n);
  n = D_ * NQ_;
  k_tcvt<<<(n + 255) / 256, 256, 0, stream>>>(Wq, WqT, D_, NQ_);
  n = D_ * LD_;
  k_tcvt<<<(n + 255) / 256, 256, 0, stream>>>(Wkvd, WkvdT, D_, LD_);
  n = LD_ * NQ_;
  k_tcvt<<<(n + 255) / 256, 256, 0, stream>>>(Wkup, WkupT, LD_, NQ_);
  k_tcvt<<<(n + 255) / 256, 256, 0, stream>>>(Wvup, WvupT, LD_, NQ_);
  n = NQ_ * D_;
  k_tcvt<<<(n + 255) / 256, 256, 0, stream>>>(Wout, WoutT, NQ_, D_);

  dim3 blk(32);  // one wave32 per block
  // q = x*Wq (scale 1/sqrt(HD) folded in), latent = x*Wkv_down
  k_gemm_nt<<<dim3(NQ_ / 64, M / 16), blk, 0, stream>>>(xb, WqT, qb, M, NQ_, D_, 0.125f, 0);
  k_gemm_nt<<<dim3(LD_ / 64, M / 16), blk, 0, stream>>>(xb, WkvdT, lat, M, LD_, D_, 1.0f, 0);
  // k = latent*Wk_up, v = latent*Wv_up (written transposed)
  k_gemm_nt<<<dim3(NQ_ / 64, M / 16), blk, 0, stream>>>(lat, WkupT, kb, M, NQ_, LD_, 1.0f, 0);
  k_gemm_nt<<<dim3(NQ_ / 64, M / 16), blk, 0, stream>>>(lat, WvupT, vT, M, NQ_, LD_, 1.0f, 1);

  // causal flash attention
  k_flash<<<dim3(L_ / 16, H_, B_), blk, 0, stream>>>(qb, kb, vT, yb);

  // out = y * Wout (fp32 output)
  k_gemm_nt<<<dim3(D_ / 64, M / 16), blk, 0, stream>>>(yb, WoutT, out, M, D_, NQ_, 1.0f, 2);
}
